// WindowAttention_18708877541684
// MI455X (gfx1250) — matrix operand order repaired and checked by reading, not compiled, not run on hardware
//
#include <hip/hip_runtime.h>
#include <hip/hip_bf16.h>
#include <stdint.h>

// ---- CDNA5 WMMA types ------------------------------------------------------
typedef __attribute__((ext_vector_type(16))) __bf16 v16bf;
typedef __attribute__((ext_vector_type(8)))  float  v8f;

union BF16x16 { v16bf v; unsigned short u[16]; uint4 q[2]; };

__device__ __forceinline__ unsigned short f2bf(float f) {
    union { float f; unsigned u; } c; c.f = f;
    unsigned u = c.u;
    u += 0x7FFFu + ((u >> 16) & 1u);          // round-to-nearest-even
    return (unsigned short)(u >> 16);
}

__device__ __forceinline__ v8f wmma_bf16(const BF16x16& a, const BF16x16& b, v8f c) {
    return __builtin_amdgcn_wmma_f32_16x16x32_bf16(false, a.v, false, b.v,
                                                   (short)0, c, false, false);
}

// Problem constants
// B=128, S=64, N=32, C=128, H=4, D=N*C/H=1024, M=B*S*N=262144
#define BHSD 33554432L   // B*H*S*D elements per Q/K/V buffer

// ---- Kernel 0: fp32 -> bf16 weight conversion ------------------------------
__global__ __launch_bounds__(256)
void convert_weights_kernel(const float* __restrict__ qkv_w, const float* __restrict__ proj_w,
                            unsigned short* __restrict__ Wqkv, unsigned short* __restrict__ Wproj) {
    int i = blockIdx.x * 256 + threadIdx.x;
    if (i < 49152) Wqkv[i]  = f2bf(qkv_w[i]);    // (384,128)
    if (i < 16384) Wproj[i] = f2bf(proj_w[i]);   // (128,128)
}

// ---- Kernel 1: QKV GEMM (262144x128 @ 128x384) + scatter into Q/K/V^T ------
// One wave per 16-row M-tile; 24 live accumulators so x is read once.
__global__ __launch_bounds__(256)
void qkv_gemm_kernel(const float* __restrict__ x, const unsigned short* __restrict__ wqkv,
                     const float* __restrict__ qkv_b,
                     unsigned short* __restrict__ Qw, unsigned short* __restrict__ Kw,
                     unsigned short* __restrict__ Vw)
{
    const int wave = threadIdx.x >> 5;
    const int lane = threadIdx.x & 31;
    const int half = lane >> 4;
    const int ln   = lane & 15;
    const int mt   = blockIdx.x * 8 + wave;
    const long row = (long)mt * 16 + ln;

    v8f acc[24];
    const v8f zero = {};
    #pragma unroll
    for (int i = 0; i < 24; ++i) acc[i] = zero;

    #pragma unroll
    for (int k0 = 0; k0 < 128; k0 += 32) {
        BF16x16 a;                                   // A 16x32: row=ln, K chunks at half*8 / 16+half*8
        const float* ap = x + row * 128 + k0;
        #pragma unroll
        for (int j = 0; j < 8; ++j) {
            a.u[j]     = f2bf(ap[half * 8 + j]);
            a.u[8 + j] = f2bf(ap[16 + half * 8 + j]);
        }
        #pragma unroll
        for (int nt = 0; nt < 24; ++nt) {
            BF16x16 bb;                              // B 32x16: col=ln, K chunks at half*8 / 16+half*8
            const unsigned short* bp = wqkv + (nt * 16 + ln) * 128 + k0 + half * 8;
            bb.q[0] = *(const uint4*)(bp);
            bb.q[1] = *(const uint4*)(bp + 16);
            acc[nt] = wmma_bf16(a, bb, acc[nt]);
        }
    }

    // Scatter: (m, n_g) -> j in {q,k,v}, head h, dim d. V stored transposed (B,H,D,S).
    // j,h are uniform over each 16-lane run, and Q/K runs are d-contiguous (coalesced).
    #pragma unroll
    for (int r = 0; r < 8; ++r) {
        const int m_g = mt * 16 + r + 8 * half;
        const int b   = m_g >> 11;
        const int s   = (m_g >> 5) & 63;
        const int nr  = m_g & 31;
        #pragma unroll
        for (int nt = 0; nt < 24; ++nt) {
            const int n_g = nt * 16 + ln;
            const unsigned short bv = f2bf(acc[nt][r] + qkv_b[n_g]);
            const int f = nr * 384 + n_g;            // position in (3,H,D)
            const int j = f >> 12;
            const int h = (f >> 10) & 3;
            const int d = f & 1023;
            if (j == 0)      Qw[(((long)b * 4 + h) * 64 + s) * 1024 + d] = bv;
            else if (j == 1) Kw[(((long)b * 4 + h) * 64 + s) * 1024 + d] = bv;
            else             Vw[(((long)b * 4 + h) * 1024 + d) * 64 + s] = bv;
        }
    }
}

// ---- Kernel 2: attention per (b,h): QK^T -> bias+softmax -> P@V ------------
__global__ __launch_bounds__(256)
void attn_kernel(const unsigned short* __restrict__ Qw, const unsigned short* __restrict__ Kw,
                 const unsigned short* __restrict__ Vw, const float* __restrict__ bias_table,
                 unsigned short* __restrict__ AO)
{
    __shared__ float sc[64][65];               // stride 65: conflict-free row softmax
    __shared__ unsigned short pr[64][64];      // bf16 probabilities (A-fragment source)
    __shared__ float btab[900];                // bias table staged via async-to-LDS

    const int bh = blockIdx.x;
    const int b = bh >> 2, h = bh & 3;
    const int wave = threadIdx.x >> 5, lane = threadIdx.x & 31;
    const int half = lane >> 4, ln = lane & 15;

    const unsigned short* Qb = Qw + (long)bh * 64 * 1024;
    const unsigned short* Kb = Kw + (long)bh * 64 * 1024;
    const unsigned short* Vb = Vw + (long)bh * 1024 * 64;   // V^T: [d][t]

    // Stage bias table (225*4 floats = 225 x 16B) into LDS asynchronously,
    // overlapped with zeroing the score accumulator.
    {
        typedef __attribute__((address_space(3))) float lds_float_t;
        const unsigned btab_base = (unsigned)(unsigned long)(lds_float_t*)&btab[0];
        if (threadIdx.x < 225) {
            const float* g = bias_table + threadIdx.x * 4;
            const unsigned ldsoff = btab_base + threadIdx.x * 16;
            asm volatile("global_load_async_to_lds_b128 %0, %1, off"
                         :: "v"(ldsoff), "v"(g) : "memory");
        }
    }
    {
        float* scf = &sc[0][0];
        for (int i = threadIdx.x; i < 64 * 65; i += 256) scf[i] = 0.f;
    }
    asm volatile("s_wait_asynccnt 0x0" ::: "memory");
    __syncthreads();

    // Phase 1: scores = Q @ K^T. Each wave owns a 2x2 tile quadrant over half
    // of the K-dim (Q and K each read twice per block); the two k-half
    // waves merge partial sums with LDS float atomics (ds_add_f32).
    {
        const int kh = wave >> 2;                 // k-half
        const int qs = (wave >> 1) & 1;           // row quadrant
        const int qt = wave & 1;                  // col quadrant
        const unsigned short* qrow0 = Qb + ((qs * 2 + 0) * 16 + ln) * 1024;
        const unsigned short* qrow1 = Qb + ((qs * 2 + 1) * 16 + ln) * 1024;
        const unsigned short* krow0 = Kb + ((qt * 2 + 0) * 16 + ln) * 1024;
        const unsigned short* krow1 = Kb + ((qt * 2 + 1) * 16 + ln) * 1024;
        v8f a00 = {}, a01 = {}, a10 = {}, a11 = {};
        const int kbeg = kh * 512;
        #pragma unroll 2
        for (int k0 = kbeg; k0 < kbeg + 512; k0 += 32) {
            BF16x16 qa0, qa1, kb0, kb1;
            qa0.q[0] = *(const uint4*)(qrow0 + k0 + half * 8);
            qa0.q[1] = *(const uint4*)(qrow0 + k0 + 16 + half * 8);
            qa1.q[0] = *(const uint4*)(qrow1 + k0 + half * 8);
            qa1.q[1] = *(const uint4*)(qrow1 + k0 + 16 + half * 8);
            kb0.q[0] = *(const uint4*)(krow0 + k0 + half * 8);
            kb0.q[1] = *(const uint4*)(krow0 + k0 + 16 + half * 8);
            kb1.q[0] = *(const uint4*)(krow1 + k0 + half * 8);
            kb1.q[1] = *(const uint4*)(krow1 + k0 + 16 + half * 8);
            a00 = wmma_bf16(qa0, kb0, a00);
            a01 = wmma_bf16(qa0, kb1, a01);
            a10 = wmma_bf16(qa1, kb0, a10);
            a11 = wmma_bf16(qa1, kb1, a11);
        }
        const int r0 = qs * 32, c0 = qt * 32;
        #pragma unroll
        for (int r = 0; r < 8; ++r) {
            const int rr = r + 8 * half;
            atomicAdd(&sc[r0 + rr][c0 + ln],           a00[r]);
            atomicAdd(&sc[r0 + rr][c0 + 16 + ln],      a01[r]);
            atomicAdd(&sc[r0 + 16 + rr][c0 + ln],      a10[r]);
            atomicAdd(&sc[r0 + 16 + rr][c0 + 16 + ln], a11[r]);
        }
    }
    __syncthreads();

    // Phase 2: bias (from LDS-staged table) + softmax, one thread per row
    if (threadIdx.x < 64) {
        const int s = threadIdx.x;
        const int ys = s >> 3, xs = s & 7;
        const float scale = 0.17677669529663687f;   // (C/H)^-0.5 = 1/sqrt(32)
        float mx = -1e30f;
        for (int t = 0; t < 64; ++t) {
            const int yt = t >> 3, xt = t & 7;
            const int ridx = (ys - yt + 7) * 15 + (xs - xt + 7);
            const float v = sc[s][t] * scale + btab[ridx * 4 + h];
            sc[s][t] = v;
            mx = fmaxf(mx, v);
        }
        float sum = 0.f;
        for (int t = 0; t < 64; ++t) { const float e = __expf(sc[s][t] - mx); sc[s][t] = e; sum += e; }
        const float inv = 1.f / sum;
        for (int t = 0; t < 64; ++t) pr[s][t] = f2bf(sc[s][t] * inv);
    }
    __syncthreads();

    // Phase 3: out = P @ V (V^T layout => contiguous B-fragments).
    // A-fragments depend only on the wave's row tile: load once, reuse for all 32 ct.
    const int rt  = wave & 3;
    const int ct0 = (wave >> 2) * 32;
    BF16x16 pa0, pa1;
    {
        const unsigned short* prow = &pr[rt * 16 + ln][0];
        pa0.q[0] = *(const uint4*)(prow + 0 + half * 8);
        pa0.q[1] = *(const uint4*)(prow + 16 + half * 8);
        pa1.q[0] = *(const uint4*)(prow + 32 + half * 8);
        pa1.q[1] = *(const uint4*)(prow + 48 + half * 8);
    }
    for (int ct = ct0; ct < ct0 + 32; ++ct) {
        v8f acc = {};
        const unsigned short* vrow = Vb + (long)(ct * 16 + ln) * 64;
        BF16x16 vb0, vb1;
        vb0.q[0] = *(const uint4*)(vrow + half * 8);
        vb0.q[1] = *(const uint4*)(vrow + 16 + half * 8);
        vb1.q[0] = *(const uint4*)(vrow + 32 + half * 8);
        vb1.q[1] = *(const uint4*)(vrow + 48 + half * 8);
        acc = wmma_bf16(pa0, vb0, acc);
        acc = wmma_bf16(pa1, vb1, acc);
        // store bf16 into (B, S*N, C) layout (= A-matrix layout for proj GEMM)
        const int d0 = ct * 16 + ln;
        const int f  = h * 1024 + d0;
        const int nr = f >> 7, c = f & 127;
        #pragma unroll
        for (int r = 0; r < 8; ++r) {
            const int s = rt * 16 + r + 8 * half;
            AO[((long)b * 2048 + s * 32 + nr) * 128 + c] = f2bf(acc[r]);
        }
    }
}

// ---- Kernel 3: proj GEMM (262144x128 @ 128x128) + bias, fp32 out -----------
__global__ __launch_bounds__(256)
void proj_gemm_kernel(const unsigned short* __restrict__ AO, const unsigned short* __restrict__ wproj,
                      const float* __restrict__ proj_b, float* __restrict__ out)
{
    const int wave = threadIdx.x >> 5, lane = threadIdx.x & 31;
    const int half = lane >> 4, ln = lane & 15;
    const int mt = blockIdx.x * 8 + wave;
    const long row = (long)mt * 16 + ln;

    v8f acc[8];
    const v8f zero = {};
    #pragma unroll
    for (int i = 0; i < 8; ++i) acc[i] = zero;

    #pragma unroll
    for (int k0 = 0; k0 < 128; k0 += 32) {
        BF16x16 a;
        const unsigned short* ap = AO + row * 128 + k0;
        a.q[0] = *(const uint4*)(ap + half * 8);
        a.q[1] = *(const uint4*)(ap + 16 + half * 8);
        #pragma unroll
        for (int nt = 0; nt < 8; ++nt) {
            BF16x16 bb;
            const unsigned short* bp = wproj + (nt * 16 + ln) * 128 + k0 + half * 8;
            bb.q[0] = *(const uint4*)(bp);
            bb.q[1] = *(const uint4*)(bp + 16);
            acc[nt] = wmma_bf16(a, bb, acc[nt]);
        }
    }
    #pragma unroll
    for (int nt = 0; nt < 8; ++nt) {
        const int n_g = nt * 16 + ln;
        const float bias = proj_b[n_g];
        #pragma unroll
        for (int r = 0; r < 8; ++r) {
            const long m_g = (long)mt * 16 + r + 8 * half;
            out[m_g * 128 + n_g] = acc[nt][r] + bias;
        }
    }
}

// ---- Host launch -----------------------------------------------------------
extern "C" void kernel_launch(void* const* d_in, const int* in_sizes, int n_in,
                              void* d_out, int out_size, void* d_ws, size_t ws_size,
                              hipStream_t stream) {
    const float* x          = (const float*)d_in[0];
    const float* qkv_w      = (const float*)d_in[1];
    const float* qkv_b      = (const float*)d_in[2];
    const float* proj_w     = (const float*)d_in[3];
    const float* proj_b     = (const float*)d_in[4];
    const float* bias_table = (const float*)d_in[5];

    unsigned short* ws    = (unsigned short*)d_ws;
    unsigned short* Qw    = ws;               // (B,H,S,D) bf16
    unsigned short* Kw    = Qw + BHSD;        // (B,H,S,D) bf16
    unsigned short* Vw    = Kw + BHSD;        // (B,H,D,S) bf16 (transposed)
    unsigned short* AO    = Vw + BHSD;        // (B,S*N,C) bf16 attention output
    unsigned short* Wqkv  = AO + BHSD;        // (384,128) bf16
    unsigned short* Wproj = Wqkv + 49152;     // (128,128) bf16

    convert_weights_kernel<<<192, 256, 0, stream>>>(qkv_w, proj_w, Wqkv, Wproj);
    qkv_gemm_kernel<<<2048, 256, 0, stream>>>(x, Wqkv, qkv_b, Qw, Kw, Vw);
    attn_kernel<<<512, 256, 0, stream>>>(Qw, Kw, Vw, bias_table, AO);
    proj_gemm_kernel<<<2048, 256, 0, stream>>>(AO, Wproj, proj_b, (float*)d_out);
}
